// CGCNN_23596550324906
// MI455X (gfx1250) — matrix operand order repaired and checked by reading, not compiled, not run on hardware
//
#include <hip/hip_runtime.h>
#include <hip/hip_bf16.h>

// ---------------------------------------------------------------------------
// CGCNN-style MPNN forward on gfx1250 (MI455X).
// All GEMMs via v_wmma_f32_16x16x32_f16 (f16 in, f32 accum).
// Weight matrices pre-shuffled into WMMA B-fragment layout each launch;
// edge kernel DMA's the fragments into LDS with the Tensor Data Mover and
// shares them across all 8 waves.  GRU+LayerNorm+pool fused per node tile.
// ---------------------------------------------------------------------------

typedef __attribute__((ext_vector_type(16))) _Float16 v16h;
typedef __attribute__((ext_vector_type(8)))  _Float16 f16x8;
typedef __attribute__((ext_vector_type(8)))  float    v8f;
typedef __attribute__((ext_vector_type(8)))  float    f32x8;

#define HID 128
#define NNODES 50000
#define NEDGES 800000
#define NGRAPHS 256
#define NCONV 3
#define WAVES_PER_BLOCK 8

#if defined(__has_builtin)
#if __has_builtin(__builtin_amdgcn_tensor_load_to_lds) && \
    __has_builtin(__builtin_amdgcn_s_wait_tensorcnt)
#define HAVE_TDM 1
#endif
#endif
#ifndef HAVE_TDM
#define HAVE_TDM 0
#endif

__device__ __forceinline__ v8f vzero8() {
    v8f z;
#pragma unroll
    for (int i = 0; i < 8; ++i) z[i] = 0.0f;
    return z;
}

// A-fragment (16x32 f16, MxK): lane holds two contiguous 8-half chunks:
// k in [kb, kb+8) and [kb+16, kb+24), kb = (lane>>4)*8, row M = lane&15.
// `src` must already point at rowBase + kt*32 + kb.
__device__ __forceinline__ v16h load_a_frag(const _Float16* src) {
    f16x8 lo = *(const f16x8*)(src);
    f16x8 hi = *(const f16x8*)(src + 16);
    v16h a;
#pragma unroll
    for (int i = 0; i < 8; ++i) { a[i] = lo[i]; a[8 + i] = hi[i]; }
    return a;
}

// Same A-fragment but gathered from an fp32 source, converted in-register.
__device__ __forceinline__ v16h load_a_frag_f32(const float* src) {
    f32x8 lo = *(const f32x8*)(src);
    f32x8 hi = *(const f32x8*)(src + 16);
    v16h a;
#pragma unroll
    for (int i = 0; i < 8; ++i) { a[i] = (_Float16)lo[i]; a[8 + i] = (_Float16)hi[i]; }
    return a;
}

__device__ __forceinline__ v8f wmma_f16(v16h a, v16h b, v8f c) {
    return __builtin_amdgcn_wmma_f32_16x16x32_f16(false, a, false, b,
                                                  (short)0, c, false, false);
}

__device__ __forceinline__ float sigmoidf_(float x) {
    return 1.0f / (1.0f + __expf(-x));
}
__device__ __forceinline__ float siluf_(float x) {
    return x / (1.0f + __expf(-x));
}

#if HAVE_TDM
typedef __attribute__((ext_vector_type(4))) unsigned int v4u;
typedef __attribute__((ext_vector_type(8))) int v8i_;
typedef __attribute__((ext_vector_type(4))) int v4i_;

__device__ __forceinline__ unsigned lds_offset(const void* p) {
    return (unsigned)(unsigned long long)(__attribute__((address_space(3))) const void*)p;
}

// 1-D DMA of `nbytes` (multiple of 4, < 256KB) from global to LDS via TDM.
__device__ __forceinline__ void tdm_load_to_lds(const void* gsrc, unsigned ldsOff,
                                                unsigned nbytes) {
    unsigned n = nbytes >> 2;  // dwords (must fit tile_dim0's 16 bits)
    unsigned long long ga = (unsigned long long)gsrc;
    v4u g0;
    g0[0] = 1u;                                   // count = 1 valid descriptor
    g0[1] = ldsOff;                               // lds_addr (bytes)
    g0[2] = (unsigned)ga;                         // global_addr[31:0]
    g0[3] = (unsigned)((ga >> 32) & 0x01FFFFFFull) | (2u << 30);  // addr[56:32], type=2
    v8i_ g1;
    g1[0] = (int)(2u << 16);                      // wg_mask=0, data_size=4B
    g1[1] = (int)(n << 16);                       // tensor_dim0 lo16 @ [63:48]
    g1[2] = (int)((n >> 16) | (1u << 16));        // tensor_dim0 hi16, tensor_dim1=1
    g1[3] = (int)(n << 16);                       // tensor_dim1 hi=0, tile_dim0=n
    g1[4] = 1;                                    // tile_dim1=1, tile_dim2=0
    g1[5] = (int)n;                               // tensor_dim0_stride lo32
    g1[6] = (int)((n & 0xFFFFu) << 16);           // stride0 hi=0, stride1 lo16
    g1[7] = (int)(n >> 16);                       // stride1 hi
    v4i_ z4; z4[0] = 0; z4[1] = 0; z4[2] = 0; z4[3] = 0;
    v8i_ z8;
#pragma unroll
    for (int i = 0; i < 8; ++i) z8[i] = 0;
    __builtin_amdgcn_tensor_load_to_lds(g0, g1, z4, z4, z8, 0);
}
#endif

// ---------------------------------------------------------------------------
// Prep kernels
// ---------------------------------------------------------------------------
__global__ void zero_kernel(float* __restrict__ p, long n) {
    long i = (long)blockIdx.x * blockDim.x + threadIdx.x;
    long stride = (long)gridDim.x * blockDim.x;
    for (; i < n; i += stride) p[i] = 0.0f;
}

__global__ void cvt_f16_kernel(const float* __restrict__ s, _Float16* __restrict__ d, long n) {
    long i = (long)blockIdx.x * blockDim.x + threadIdx.x;
    long stride = (long)gridDim.x * blockDim.x;
    for (; i < n; i += stride) d[i] = (_Float16)s[i];
}

// Shuffle W[K][N] (row stride ldw floats) into B-fragment layout:
// frag tile (kt,nt) -> 32 lanes x 16 halves; half t of lane l is
// W[kt*32 + (l>>4)*8 + (t&7) + (t>>3)*16][nt*16 + (l&15)].
__global__ void shuffle_b_kernel(const float* __restrict__ W, _Float16* __restrict__ dst,
                                 int KT, int NT, int ldw) {
    long total = (long)KT * NT * 512;
    long i = (long)blockIdx.x * blockDim.x + threadIdx.x;
    long stride = (long)gridDim.x * blockDim.x;
    for (; i < total; i += stride) {
        int t    = (int)(i & 15);
        int lane = (int)((i >> 4) & 31);
        long tile = i >> 9;
        int nt = (int)(tile % NT);
        int kt = (int)(tile / NT);
        int k = kt * 32 + (lane >> 4) * 8 + (t & 7) + (t >> 3) * 16;
        int n = nt * 16 + (lane & 15);
        dst[i] = (_Float16)W[(long)k * ldw + n];
    }
}

// ---------------------------------------------------------------------------
// Node init: h = emb[x] @ proj_w + proj_b   (one wave per 16-node tile)
// ---------------------------------------------------------------------------
__global__ __launch_bounds__(256) void node_init_kernel(
    const int* __restrict__ x, const _Float16* __restrict__ emb16,
    const _Float16* __restrict__ pwf, const float* __restrict__ pb,
    float* __restrict__ h32, _Float16* __restrict__ h16, int N) {
    int lane = threadIdx.x & 31;
    int wave = threadIdx.x >> 5;
    long tile = (long)blockIdx.x * WAVES_PER_BLOCK + wave;
    long base = tile * 16;
    if (base >= N) return;
    int rA = lane & 15;
    int kb = (lane >> 4) * 8;
    int hb = lane >> 4;

    long nA = base + rA; if (nA > N - 1) nA = N - 1;
    const _Float16* er = emb16 + (long)x[nA] * HID;

    v16h aF[4];
#pragma unroll
    for (int kt = 0; kt < 4; ++kt) aF[kt] = load_a_frag(er + kt * 32 + kb);

    v8f acc[8];
#pragma unroll
    for (int nt = 0; nt < 8; ++nt) acc[nt] = vzero8();
#pragma unroll
    for (int nt = 0; nt < 8; ++nt) {
#pragma unroll
        for (int kt = 0; kt < 4; ++kt) {
            v16h b = *(const v16h*)(pwf + ((long)(kt * 8 + nt) * 32 + lane) * 16);
            acc[nt] = wmma_f16(aF[kt], b, acc[nt]);
        }
    }
    int colL = lane & 15;
#pragma unroll
    for (int nt = 0; nt < 8; ++nt) {
        int col = nt * 16 + colL;
        float bb = pb[col];
#pragma unroll
        for (int v = 0; v < 8; ++v) {
            long node = base + hb * 8 + v;
            if (node < N) {
                float val = acc[nt][v] + bb;
                h32[node * HID + col] = val;
                h16[node * HID + col] = (_Float16)val;
            }
        }
    }
}

// ---------------------------------------------------------------------------
// Edge message MLP + scatter-add.  One wave per 16-edge tile.
// Weight fragments DMA'd to LDS (TDM) and shared by all waves in the block.
// GEMM1: [16x256] (gathered h[row]||h[col], f16) @ W1 -> SiLU -> LDS staging
// GEMM2: [16x128] @ W2 -> atomic add into agg[row].
// edge_attr column (k=256 of W1) folded into the bias per-edge.
// ---------------------------------------------------------------------------
__global__ __launch_bounds__(256) void edge_msg_kernel(
    const _Float16* __restrict__ h16, const int* __restrict__ eidx,
    const float* __restrict__ eattr,
    const _Float16* __restrict__ w1f, const float* __restrict__ w1last,
    const float* __restrict__ b1,
    const _Float16* __restrict__ w2f, const float* __restrict__ b2,
    float* __restrict__ agg, int E) {
    // 64KB of W1 fragments (later re-used as SiLU staging) + 32KB of W2 frags.
    __shared__ __align__(32) _Float16 ldsW1[8 * 8 * 512];
    __shared__ __align__(32) _Float16 ldsW2[4 * 8 * 512];
    const int tid = threadIdx.x;
    int lane = tid & 31;
    int wave = tid >> 5;

#if HAVE_TDM
    if (tid < 32) {
        tdm_load_to_lds(w1f, lds_offset(ldsW1), 8 * 8 * 512 * 2);
        tdm_load_to_lds(w2f, lds_offset(ldsW2), 4 * 8 * 512 * 2);
        __builtin_amdgcn_s_wait_tensorcnt(0);
    }
#else
    for (int i = tid; i < 8 * 8 * 512 / 8; i += 256)
        ((f16x8*)ldsW1)[i] = ((const f16x8*)w1f)[i];
    for (int i = tid; i < 4 * 8 * 512 / 8; i += 256)
        ((f16x8*)ldsW2)[i] = ((const f16x8*)w2f)[i];
#endif
    __syncthreads();

    long tile = (long)blockIdx.x * WAVES_PER_BLOCK + wave;
    long e0 = tile * 16;
    const int* rowp = eidx;
    const int* colp = eidx + E;
    int rA = lane & 15;
    int kb = (lane >> 4) * 8;
    int hb = lane >> 4;

    long eA = e0 + rA; if (eA > E - 1) eA = E - 1;
    const _Float16* hr = h16 + (long)rowp[eA] * HID;
    const _Float16* hc = h16 + (long)colp[eA] * HID;

    // preload all 8 A fragments (gathered node features)
    v16h aF[8];
#pragma unroll
    for (int kt = 0; kt < 8; ++kt) {
        const _Float16* src = (kt < 4) ? (hr + kt * 32 + kb) : (hc + (kt - 4) * 32 + kb);
        aF[kt] = load_a_frag(src);
    }
    v8f acc[8];
#pragma unroll
    for (int nt = 0; nt < 8; ++nt) acc[nt] = vzero8();
#pragma unroll
    for (int nt = 0; nt < 8; ++nt) {
#pragma unroll
        for (int kt = 0; kt < 8; ++kt) {
            v16h b = *(const v16h*)(ldsW1 + ((kt * 8 + nt) * 32 + lane) * 16);
            acc[nt] = wmma_f16(aF[kt], b, acc[nt]);
        }
    }
    float we[8];
#pragma unroll
    for (int v = 0; v < 8; ++v) {
        long ee = e0 + hb * 8 + v; if (ee > E - 1) ee = E - 1;
        we[v] = eattr[ee];
    }
    __syncthreads();   // everyone done reading W1 fragments from LDS
    // epilogue 1: + b1 + w_e * W1[256,:], SiLU, stage f16 tile into LDS
    _Float16* myS = ldsW1 + wave * (16 * HID);
    int colL = lane & 15;
#pragma unroll
    for (int nt = 0; nt < 8; ++nt) {
        int col = nt * 16 + colL;
        float bb = b1[col];
        float wl = w1last[col];
#pragma unroll
        for (int v = 0; v < 8; ++v) {
            float val = acc[nt][v] + bb + we[v] * wl;
            myS[(hb * 8 + v) * HID + col] = (_Float16)siluf_(val);
        }
    }
    __syncthreads();
    // GEMM2: hidden(16x128) @ W2
    v16h aS[4];
#pragma unroll
    for (int kt = 0; kt < 4; ++kt)
        aS[kt] = load_a_frag(myS + rA * HID + kt * 32 + kb);
    v8f acc2[8];
#pragma unroll
    for (int nt = 0; nt < 8; ++nt) acc2[nt] = vzero8();
#pragma unroll
    for (int nt = 0; nt < 8; ++nt) {
#pragma unroll
        for (int kt = 0; kt < 4; ++kt) {
            v16h b = *(const v16h*)(ldsW2 + ((kt * 8 + nt) * 32 + lane) * 16);
            acc2[nt] = wmma_f16(aS[kt], b, acc2[nt]);
        }
    }
    // epilogue 2: + b2, scatter-add to agg[row[e]]
    int rdst[8];
#pragma unroll
    for (int v = 0; v < 8; ++v) {
        long ee = e0 + hb * 8 + v; if (ee > E - 1) ee = E - 1;
        rdst[v] = rowp[ee];
    }
#pragma unroll
    for (int nt = 0; nt < 8; ++nt) {
        int col = nt * 16 + colL;
        float bb = b2[col];
#pragma unroll
        for (int v = 0; v < 8; ++v) {
            long ee = e0 + hb * 8 + v;
            if (ee < E)
                atomicAdd(&agg[(long)rdst[v] * HID + col], acc2[nt][v] + bb);
        }
    }
}

// ---------------------------------------------------------------------------
// Fused GRU + residual + LayerNorm (+ optional mean-pool accumulate).
// One wave per 16-node tile.  agg read as fp32 (converted in-register),
// full 16x128 output tile kept in registers, mean/var reduced with wave32
// shuffles across the 16-lane column groups.
// ---------------------------------------------------------------------------
__global__ __launch_bounds__(256) void gru_ln_kernel(
    const float* __restrict__ agg, const _Float16* __restrict__ h16in,
    float* __restrict__ h32, _Float16* __restrict__ h16out,
    const _Float16* __restrict__ wihf, const _Float16* __restrict__ whhf,
    const float* __restrict__ bih, const float* __restrict__ bhh,
    const float* __restrict__ lng, const float* __restrict__ lnb,
    const int* __restrict__ batch, float* __restrict__ sums,
    float* __restrict__ cnt, int doPool, int N) {
    int lane = threadIdx.x & 31;
    int wave = threadIdx.x >> 5;
    long tile = (long)blockIdx.x * WAVES_PER_BLOCK + wave;
    long base = tile * 16;
    if (base >= N) return;
    int rA = lane & 15;
    int kb = (lane >> 4) * 8;
    int hb = lane >> 4;
    long nA = base + rA; if (nA > N - 1) nA = N - 1;
    const float*    arp = agg + nA * HID;
    const _Float16* hpp = h16in + nA * HID;
    int colL = lane & 15;

    v16h aA[4], aH[4];
#pragma unroll
    for (int kt = 0; kt < 4; ++kt) {
        aA[kt] = load_a_frag_f32(arp + kt * 32 + kb);
        aH[kt] = load_a_frag(hpp + kt * 32 + kb);
    }

    float out[64];          // [nt*8+v] GRU output (h + h_new), registers
    float srow[8], ssrow[8];
#pragma unroll
    for (int v = 0; v < 8; ++v) { srow[v] = 0.0f; ssrow[v] = 0.0f; }

#pragma unroll
    for (int nt = 0; nt < 8; ++nt) {
        v8f ir = vzero8(), iz = vzero8(), in_ = vzero8();
        v8f hr_ = vzero8(), hz = vzero8(), hn_ = vzero8();
#pragma unroll
        for (int kt = 0; kt < 4; ++kt) {
            v16h bi, bh;
            bi = *(const v16h*)(wihf + ((long)(kt * 24 + nt) * 32 + lane) * 16);
            bh = *(const v16h*)(whhf + ((long)(kt * 24 + nt) * 32 + lane) * 16);
            ir  = wmma_f16(aA[kt], bi, ir);
            hr_ = wmma_f16(aH[kt], bh, hr_);
            bi = *(const v16h*)(wihf + ((long)(kt * 24 + nt + 8) * 32 + lane) * 16);
            bh = *(const v16h*)(whhf + ((long)(kt * 24 + nt + 8) * 32 + lane) * 16);
            iz = wmma_f16(aA[kt], bi, iz);
            hz = wmma_f16(aH[kt], bh, hz);
            bi = *(const v16h*)(wihf + ((long)(kt * 24 + nt + 16) * 32 + lane) * 16);
            bh = *(const v16h*)(whhf + ((long)(kt * 24 + nt + 16) * 32 + lane) * 16);
            in_ = wmma_f16(aA[kt], bi, in_);
            hn_ = wmma_f16(aH[kt], bh, hn_);
        }
        int col = nt * 16 + colL;
        float bir = bih[col], biz = bih[HID + col], bin = bih[2 * HID + col];
        float bhr = bhh[col], bhz = bhh[HID + col], bhn = bhh[2 * HID + col];
#pragma unroll
        for (int v = 0; v < 8; ++v) {
            long node = base + hb * 8 + v;
            long nc = (node < N ? node : (long)N - 1) * HID + col;
            float r = sigmoidf_((ir[v] + bir) + (hr_[v] + bhr));
            float z = sigmoidf_((iz[v] + biz) + (hz[v] + bhz));
            float hv = h32[nc];
            float n = tanhf((in_[v] + bin) + r * (hn_[v] + bhn));
            float o = hv + ((1.0f - z) * n + z * hv);
            out[nt * 8 + v] = o;
            srow[v] += o;
            ssrow[v] += o * o;
        }
    }
    // reduce mean / var across the 16 column lanes of each half-wave
#pragma unroll
    for (int v = 0; v < 8; ++v) {
#pragma unroll
        for (int off = 8; off > 0; off >>= 1) {
            srow[v]  += __shfl_xor(srow[v],  off, 32);
            ssrow[v] += __shfl_xor(ssrow[v], off, 32);
        }
    }
    float mu[8], inv[8];
#pragma unroll
    for (int v = 0; v < 8; ++v) {
        mu[v] = srow[v] * (1.0f / HID);
        float var = ssrow[v] * (1.0f / HID) - mu[v] * mu[v];
        inv[v] = rsqrtf(var + 1e-5f);
    }
    int gidx[8];
    if (doPool) {
#pragma unroll
        for (int v = 0; v < 8; ++v) {
            long node = base + hb * 8 + v;
            gidx[v] = batch[node < N ? node : (long)N - 1];
        }
    }
#pragma unroll
    for (int nt = 0; nt < 8; ++nt) {
        int col = nt * 16 + colL;
        float gc = lng[col], bc = lnb[col];
#pragma unroll
        for (int v = 0; v < 8; ++v) {
            long node = base + hb * 8 + v;
            if (node < N) {
                float y = (out[nt * 8 + v] - mu[v]) * inv[v] * gc + bc;
                h32[node * HID + col] = y;
                h16out[node * HID + col] = (_Float16)y;
                if (doPool) atomicAdd(&sums[(long)gidx[v] * HID + col], y);
            }
        }
    }
    if (doPool && colL == 0) {
#pragma unroll
        for (int v = 0; v < 8; ++v) {
            long node = base + hb * 8 + v;
            if (node < N) atomicAdd(&cnt[gidx[v]], 1.0f);
        }
    }
}

// ---------------------------------------------------------------------------
// Head MLP over pooled graph vectors
// ---------------------------------------------------------------------------
__global__ __launch_bounds__(128) void head_kernel(
    const float* __restrict__ sums, const float* __restrict__ cnt,
    const float* __restrict__ w1, const float* __restrict__ b1,
    const float* __restrict__ w2, const float* __restrict__ b2,
    float* __restrict__ out) {
    __shared__ float gvec[HID];
    __shared__ float red[HID];
    int bg = blockIdx.x;
    int t = threadIdx.x;
    float c = cnt[bg];
    if (c < 1.0f) c = 1.0f;
    gvec[t] = sums[(long)bg * HID + t] / c;
    __syncthreads();
    float acc = 0.0f;
#pragma unroll 8
    for (int k = 0; k < HID; ++k) acc += gvec[k] * w1[(long)k * HID + t];
    acc += b1[t];
    red[t] = siluf_(acc) * w2[t];
    __syncthreads();
#pragma unroll
    for (int off = 64; off > 0; off >>= 1) {
        if (t < off) red[t] += red[t + off];
        __syncthreads();
    }
    if (t == 0) out[bg] = red[0] + b2[0];
}

// ---------------------------------------------------------------------------
// Host launcher
// ---------------------------------------------------------------------------
extern "C" void kernel_launch(void* const* d_in, const int* in_sizes, int n_in,
                              void* d_out, int out_size, void* d_ws, size_t ws_size,
                              hipStream_t stream) {
    (void)in_sizes; (void)n_in; (void)out_size; (void)ws_size;
    const int N = NNODES, E = NEDGES, G = NGRAPHS;

    const int*   x        = (const int*)d_in[0];
    const int*   eidx     = (const int*)d_in[1];
    const float* eattr    = (const float*)d_in[2];
    const int*   batch    = (const int*)d_in[3];
    const float* emb      = (const float*)d_in[4];
    const float* proj_w   = (const float*)d_in[5];
    const float* proj_b   = (const float*)d_in[6];
    const float* msg_w1   = (const float*)d_in[7];
    const float* msg_b1   = (const float*)d_in[8];
    const float* msg_w2   = (const float*)d_in[9];
    const float* msg_b2   = (const float*)d_in[10];
    const float* gru_wih  = (const float*)d_in[11];
    const float* gru_whh  = (const float*)d_in[12];
    const float* gru_bih  = (const float*)d_in[13];
    const float* gru_bhh  = (const float*)d_in[14];
    const float* ln_g     = (const float*)d_in[15];
    const float* ln_b     = (const float*)d_in[16];
    const float* head_w1  = (const float*)d_in[17];
    const float* head_b1  = (const float*)d_in[18];
    const float* head_w2  = (const float*)d_in[19];
    const float* head_b2  = (const float*)d_in[20];
    float* out = (float*)d_out;

    char* p = (char*)d_ws;
    auto alloc = [&](size_t bytes) -> char* {
        char* r = p;
        p += (bytes + 255) & ~(size_t)255;
        return r;
    };
    float*     h32    = (float*)alloc((size_t)N * HID * 4);
    float*     agg    = (float*)alloc((size_t)N * HID * 4);
    _Float16*  h16    = (_Float16*)alloc((size_t)N * HID * 2);
    _Float16*  emb16  = (_Float16*)alloc((size_t)101 * HID * 2);
    _Float16*  projwF = (_Float16*)alloc((size_t)4 * 8 * 512 * 2);
    _Float16*  w1F    = (_Float16*)alloc((size_t)NCONV * 8 * 8 * 512 * 2);
    _Float16*  w2F    = (_Float16*)alloc((size_t)NCONV * 4 * 8 * 512 * 2);
    _Float16*  wihF   = (_Float16*)alloc((size_t)NCONV * 4 * 24 * 512 * 2);
    _Float16*  whhF   = (_Float16*)alloc((size_t)NCONV * 4 * 24 * 512 * 2);
    float*     sums   = (float*)alloc((size_t)G * HID * 4);
    float*     cnt    = (float*)alloc((size_t)G * 4);

    // ---- weight prep (every launch; cheap, deterministic) ----
    cvt_f16_kernel<<<64, 256, 0, stream>>>(emb, emb16, (long)101 * HID);
    shuffle_b_kernel<<<64, 256, 0, stream>>>(proj_w, projwF, 4, 8, HID);
    for (int l = 0; l < NCONV; ++l) {
        shuffle_b_kernel<<<128, 256, 0, stream>>>(msg_w1 + (long)l * 257 * HID,
                                                  w1F + (long)l * 8 * 8 * 512, 8, 8, HID);
        shuffle_b_kernel<<<64, 256, 0, stream>>>(msg_w2 + (long)l * HID * HID,
                                                 w2F + (long)l * 4 * 8 * 512, 4, 8, HID);
        shuffle_b_kernel<<<192, 256, 0, stream>>>(gru_wih + (long)l * HID * 384,
                                                  wihF + (long)l * 4 * 24 * 512, 4, 24, 384);
        shuffle_b_kernel<<<192, 256, 0, stream>>>(gru_whh + (long)l * HID * 384,
                                                  whhF + (long)l * 4 * 24 * 512, 4, 24, 384);
    }

    // ---- node embedding + projection ----
    int nodeTiles = (N + 15) / 16;
    int nodeBlocks = (nodeTiles + WAVES_PER_BLOCK - 1) / WAVES_PER_BLOCK;
    node_init_kernel<<<nodeBlocks, 256, 0, stream>>>(x, emb16, projwF, proj_b, h32, h16, N);

    // ---- pooling accumulators (written by last fused GRU/LN layer) ----
    zero_kernel<<<128, 256, 0, stream>>>(sums, (long)G * HID);
    zero_kernel<<<1, 256, 0, stream>>>(cnt, (long)G);

    // ---- conv layers ----
    int edgeTiles = (E + 15) / 16;
    int edgeBlocks = (edgeTiles + WAVES_PER_BLOCK - 1) / WAVES_PER_BLOCK;
    for (int l = 0; l < NCONV; ++l) {
        zero_kernel<<<4096, 256, 0, stream>>>(agg, (long)N * HID);
        edge_msg_kernel<<<edgeBlocks, 256, 0, stream>>>(
            h16, eidx, eattr,
            w1F + (long)l * 8 * 8 * 512,
            msg_w1 + (long)l * 257 * HID + 256 * HID,
            msg_b1 + (long)l * HID,
            w2F + (long)l * 4 * 8 * 512,
            msg_b2 + (long)l * HID,
            agg, E);
        gru_ln_kernel<<<nodeBlocks, 256, 0, stream>>>(
            agg, h16, h32, h16,
            wihF + (long)l * 4 * 24 * 512,
            whhF + (long)l * 4 * 24 * 512,
            gru_bih + (long)l * 3 * HID,
            gru_bhh + (long)l * 3 * HID,
            ln_g + (long)l * HID,
            ln_b + (long)l * HID,
            batch, sums, cnt, (l == NCONV - 1) ? 1 : 0, N);
    }

    // ---- head ----
    head_kernel<<<G, 128, 0, stream>>>(sums, cnt, head_w1, head_b1, head_w2, head_b2, out);
}
